// TransformerLayer_78486232367471
// MI455X (gfx1250) — matrix operand order repaired and checked by reading, not compiled, not run on hardware
//
#include <hip/hip_runtime.h>
#include <hip/hip_bf16.h>

// ---------------------------------------------------------------- constants
constexpr int CB  = 4;      // batch
constexpr int CS  = 2048;   // sequence
constexpr int CH  = 8;      // heads
constexpr int CD  = 768;    // d_model
constexpr int CF  = 3072;   // d_ff
constexpr int CDh = 96;     // head dim
constexpr int CNT = CB * CS;          // 8192 tokens

typedef unsigned short u16;
typedef __attribute__((ext_vector_type(16))) __bf16 v16bf;
typedef __attribute__((ext_vector_type(8)))  __bf16 v8bf;
typedef __attribute__((ext_vector_type(8)))  float  v8f;

// ---------------------------------------------------------------- helpers
__device__ __forceinline__ u16 f2bf(float f) {
    unsigned int u = __float_as_uint(f);
    u += 0x7FFFu + ((u >> 16) & 1u);   // round-to-nearest-even
    return (u16)(u >> 16);
}

__device__ __forceinline__ v8bf ld8(const u16* p) {
    return *reinterpret_cast<const v8bf*>(p);
}

__device__ __forceinline__ v16bf mk16(v8bf lo, v8bf hi) {
    union U { struct { v8bf a, b; } s; v16bf v; } u;
    u.s.a = lo; u.s.b = hi;
    return u.v;
}

__device__ __forceinline__ v8f v8f_zero() {
    v8f z = {0.f, 0.f, 0.f, 0.f, 0.f, 0.f, 0.f, 0.f};
    return z;
}

__device__ __forceinline__ v8f wmma_bf16(v16bf a, v16bf b, v8f c) {
    return __builtin_amdgcn_wmma_f32_16x16x32_bf16(
        /*neg_a=*/false, a, /*neg_b=*/false, b,
        /*c_mod=*/(short)0, c, /*reuse_a=*/false, /*reuse_b=*/false);
}

// gfx1250 async direct global->LDS copy (16B per lane), tracked by ASYNCcnt.
__device__ __forceinline__ void async_ld16(void* lds, const void* g) {
    unsigned l = (unsigned)(size_t)lds;     // low 32 bits of LDS aperture addr = LDS offset
    asm volatile("global_load_async_to_lds_b128 %0, %1, off"
                 :: "v"(l), "v"(g) : "memory");
}
__device__ __forceinline__ void wait_async0() {
    asm volatile("s_wait_asynccnt 0" ::: "memory");
}

// ---------------------------------------------------------------- fp32 -> bf16
__global__ void k_f32_to_bf16(const float* __restrict__ x, u16* __restrict__ y, int n) {
    int i = blockIdx.x * 256 + threadIdx.x;
    if (i < n) y[i] = f2bf(x[i]);
}

// ---------------------------------------------------------------- generic WMMA GEMM
// C[M,N] = A[M,K](bf16) * Bt[N,K](bf16)^T + bias
// MODE 0: write fp32 C            MODE 1: gelu -> write bf16 Cb
// MODE 2: += residual R, write fp32 C
#define BKK 32
#define LDT 40   // padded LDS stride (bf16 elems): 80B rows, conflict-free frags

template <int MODE>
__global__ __launch_bounds__(256)
void k_gemm(const u16* __restrict__ A, const u16* __restrict__ Bt,
            const float* __restrict__ bias, const float* __restrict__ R,
            float* __restrict__ Cf, u16* __restrict__ Cb,
            int M, int N, int K)
{
    __shared__ u16 As[2][128 * LDT];
    __shared__ u16 Bs[2][128 * LDT];

    const int tid  = threadIdx.x;
    const int lane = tid & 31;
    const int w    = tid >> 5;       // 0..7
    const int wrow = w >> 1;         // 0..3  -> 32 rows each
    const int wcol = w & 1;          // 0..1  -> 64 cols each
    const int l15  = lane & 15;
    const int hi   = lane >> 4;
    const int m0   = blockIdx.y * 128;
    const int n0   = blockIdx.x * 128;

    v8f acc[2][4];
#pragma unroll
    for (int i = 0; i < 2; i++)
#pragma unroll
        for (int j = 0; j < 4; j++) acc[i][j] = v8f_zero();

    const int r0 = tid >> 2;          // 0..63
    const int cg = (tid & 3) * 8;     // 0,8,16,24

    auto issue_tile = [&](int buf, int kt) {
        const u16* ga = A + (size_t)(m0 + r0) * K + kt + cg;
        async_ld16(&As[buf][r0 * LDT + cg],        ga);
        async_ld16(&As[buf][(r0 + 64) * LDT + cg], ga + (size_t)64 * K);
        const u16* gb = Bt + (size_t)(n0 + r0) * K + kt + cg;
        async_ld16(&Bs[buf][r0 * LDT + cg],        gb);
        async_ld16(&Bs[buf][(r0 + 64) * LDT + cg], gb + (size_t)64 * K);
    };

    issue_tile(0, 0);
    int cur = 0;
    for (int kt = 0; kt < K; kt += BKK) {
        wait_async0();       // this wave's loads for buf 'cur' are in LDS
        __syncthreads();     // everyone's loads done; prev buffer free for overwrite
        if (kt + BKK < K) issue_tile(cur ^ 1, kt + BKK);

        v16bf af[2], bfm[4];
#pragma unroll
        for (int mt = 0; mt < 2; mt++) {
            int row = wrow * 32 + mt * 16 + l15;
            af[mt] = mk16(ld8(&As[cur][row * LDT + hi * 8]),
                          ld8(&As[cur][row * LDT + 16 + hi * 8]));
        }
#pragma unroll
        for (int nt = 0; nt < 4; nt++) {
            int row = wcol * 64 + nt * 16 + l15;
            bfm[nt] = mk16(ld8(&Bs[cur][row * LDT + hi * 8]),
                           ld8(&Bs[cur][row * LDT + 16 + hi * 8]));
        }
#pragma unroll
        for (int mt = 0; mt < 2; mt++)
#pragma unroll
            for (int nt = 0; nt < 4; nt++)
                acc[mt][nt] = wmma_bf16(af[mt], bfm[nt], acc[mt][nt]);
        cur ^= 1;
    }

#pragma unroll
    for (int mt = 0; mt < 2; mt++) {
#pragma unroll
        for (int nt = 0; nt < 4; nt++) {
            int col   = n0 + wcol * 64 + nt * 16 + l15;
            float bv  = bias[col];
#pragma unroll
            for (int r = 0; r < 8; r++) {
                int row = m0 + wrow * 32 + mt * 16 + hi * 8 + r;
                float v = acc[mt][nt][r] + bv;
                size_t idx = (size_t)row * N + col;
                if (MODE == 0) {
                    Cf[idx] = v;
                } else if (MODE == 1) {
                    float g = 0.5f * v * (1.f + erff(v * 0.70710678118654752f));
                    Cb[idx] = f2bf(g);
                } else {
                    Cf[idx] = v + R[idx];
                }
            }
        }
    }
}

// ---------------------------------------------------------------- RoPE3D -> bf16 heads
__global__ void k_rope(const float* __restrict__ Qr, const float* __restrict__ Kr,
                       const float* __restrict__ coords,
                       u16* __restrict__ Qh, u16* __restrict__ Kh)
{
    size_t i = (size_t)blockIdx.x * 256 + threadIdx.x;
    const size_t total = (size_t)CB * CS * CH * 48;
    if (i >= total) return;
    int pr  = (int)(i % 48);
    size_t t = i / 48;
    int h   = (int)(t % CH);
    size_t bs = t / CH;                   // b*S + s
    int band  = pr / 3;
    int coord = pr % 3;
    float theta = __powf(10000.f, -(float)band * (1.f / 16.f));
    float ang   = coords[bs * 3 + coord] * theta;
    float sn, cs;
    __sincosf(ang, &sn, &cs);
    int d0 = band * 6 + coord * 2;
    size_t inIdx = bs * CD + (size_t)h * CDh + d0;
    float x0q = Qr[inIdx], x1q = Qr[inIdx + 1];
    float x0k = Kr[inIdx], x1k = Kr[inIdx + 1];
    int b = (int)(bs / CS), s = (int)(bs % CS);
    size_t outIdx = (((size_t)(b * CH + h)) * CS + s) * CDh + d0;
    Qh[outIdx]     = f2bf(x0q * cs - x1q * sn);
    Qh[outIdx + 1] = f2bf(x0q * sn + x1q * cs);
    Kh[outIdx]     = f2bf(x0k * cs - x1k * sn);
    Kh[outIdx + 1] = f2bf(x0k * sn + x1k * cs);
}

// ---------------------------------------------------------------- V reshape -> [B,H,Dh,S] bf16
__global__ void k_vre(const float* __restrict__ Vr, u16* __restrict__ Vt)
{
    size_t i = (size_t)blockIdx.x * 256 + threadIdx.x;
    const size_t total = (size_t)CNT * CD;
    if (i >= total) return;
    int c = (int)(i % CD);
    size_t bs = i / CD;
    int s = (int)(bs % CS), b = (int)(bs / CS);
    int h = c / CDh, d = c % CDh;
    Vt[(((size_t)(b * CH + h)) * CDh + d) * CS + s] = f2bf(Vr[i]);
}

// ---------------------------------------------------------------- flash attention
// grid (S/64, B*H), 128 threads (4 waves); wave w owns queries [w*16, w*16+16)
#define LDKT 104   // Ks padded stride (bf16)
#define LDVT 72    // Vs padded stride
#define LDPT 72    // Pb padded stride

__global__ __launch_bounds__(128)
void k_attn(const u16* __restrict__ Qh, const u16* __restrict__ Kh,
            const u16* __restrict__ Vt, u16* __restrict__ CtxB, float scale)
{
    __shared__ u16 Ks[2][64 * LDKT];
    __shared__ u16 Vs[2][96 * LDVT];
    __shared__ u16 Pb[64 * LDPT];

    const int tid  = threadIdx.x;
    const int lane = tid & 31;
    const int wv   = tid >> 5;     // 0..3
    const int l15  = lane & 15;
    const int hi   = lane >> 4;
    const int bh   = blockIdx.y;   // b*H + h
    const int b    = bh / CH, h = bh % CH;
    const int q0   = blockIdx.x * 64;

    // Q fragments (3 k-steps of 32) straight from global [B,H,S,96]
    const u16* qrow = Qh + ((size_t)bh * CS + q0 + wv * 16 + l15) * CDh;
    v16bf qf[3];
#pragma unroll
    for (int ks = 0; ks < 3; ks++)
        qf[ks] = mk16(ld8(qrow + ks * 32 + hi * 8),
                      ld8(qrow + ks * 32 + 16 + hi * 8));

    v8f ctx[6];
#pragma unroll
    for (int i = 0; i < 6; i++) ctx[i] = v8f_zero();
    float mrun[8], lrun[8];
#pragma unroll
    for (int r = 0; r < 8; r++) { mrun[r] = -1e30f; lrun[r] = 0.f; }

    auto issue_kv = [&](int buf, int kb) {
        const u16* kbase = Kh + ((size_t)bh * CS + kb * 64) * CDh;
        for (int c = tid; c < 768; c += 128) {
            int row = c / 12, cgk = (c % 12) * 8;
            async_ld16(&Ks[buf][row * LDKT + cgk], kbase + (size_t)row * CDh + cgk);
        }
        const u16* vbase = Vt + (size_t)bh * CDh * CS + kb * 64;
        for (int c = tid; c < 768; c += 128) {
            int row = c / 8, cgv = (c % 8) * 8;
            async_ld16(&Vs[buf][row * LDVT + cgv], vbase + (size_t)row * CS + cgv);
        }
    };

    issue_kv(0, 0);
    int cur = 0;
    const int nkb = CS / 64;
    for (int kb = 0; kb < nkb; kb++) {
        wait_async0();
        __syncthreads();
        if (kb + 1 < nkb) issue_kv(cur ^ 1, kb + 1);

        // scores: 4 N-tiles of 16 keys, Dh=96 -> 3 wmma k-steps each
        v8f sc[4];
#pragma unroll
        for (int nt = 0; nt < 4; nt++) {
            v8f a = v8f_zero();
#pragma unroll
            for (int ks = 0; ks < 3; ks++) {
                int krow = nt * 16 + l15;
                a = wmma_bf16(qf[ks],
                              mk16(ld8(&Ks[cur][krow * LDKT + ks * 32 + hi * 8]),
                                   ld8(&Ks[cur][krow * LDKT + ks * 32 + 16 + hi * 8])),
                              a);
            }
            sc[nt] = a;
        }

        // online softmax, fully in-register (row = wv*16 + hi*8 + r)
        float al[8];
#pragma unroll
        for (int r = 0; r < 8; r++) {
            float s0 = sc[0][r] * scale, s1 = sc[1][r] * scale;
            float s2 = sc[2][r] * scale, s3 = sc[3][r] * scale;
            float bm = fmaxf(fmaxf(s0, s1), fmaxf(s2, s3));
#pragma unroll
            for (int off = 1; off < 16; off <<= 1)
                bm = fmaxf(bm, __shfl_xor(bm, off, 32));
            float nm = fmaxf(mrun[r], bm);
            float p0 = __expf(s0 - nm), p1 = __expf(s1 - nm);
            float p2 = __expf(s2 - nm), p3 = __expf(s3 - nm);
            float ps = p0 + p1 + p2 + p3;
#pragma unroll
            for (int off = 1; off < 16; off <<= 1)
                ps += __shfl_xor(ps, off, 32);
            al[r]   = __expf(mrun[r] - nm);
            lrun[r] = lrun[r] * al[r] + ps;
            mrun[r] = nm;
            int prow = wv * 16 + hi * 8 + r;
            Pb[prow * LDPT +  0 + l15] = f2bf(p0);
            Pb[prow * LDPT + 16 + l15] = f2bf(p1);
            Pb[prow * LDPT + 32 + l15] = f2bf(p2);
            Pb[prow * LDPT + 48 + l15] = f2bf(p3);
        }
        __syncthreads();

        // rescale accumulators by alpha
#pragma unroll
        for (int t = 0; t < 6; t++)
#pragma unroll
            for (int r = 0; r < 8; r++) ctx[t][r] *= al[r];

        // ctx += P(16x64) @ V(64x96) : A = Pb, B^T = Vs
#pragma unroll
        for (int ks = 0; ks < 2; ks++) {
            int arow = wv * 16 + l15;
            v16bf pa = mk16(ld8(&Pb[arow * LDPT + ks * 32 + hi * 8]),
                            ld8(&Pb[arow * LDPT + ks * 32 + 16 + hi * 8]));
#pragma unroll
            for (int nt = 0; nt < 6; nt++) {
                int vrow = nt * 16 + l15;
                ctx[nt] = wmma_bf16(pa,
                                    mk16(ld8(&Vs[cur][vrow * LDVT + ks * 32 + hi * 8]),
                                         ld8(&Vs[cur][vrow * LDVT + ks * 32 + 16 + hi * 8])),
                                    ctx[nt]);
            }
        }
        cur ^= 1;
    }

    // normalize + store bf16 ctx into [B*S, 768]
#pragma unroll
    for (int nt = 0; nt < 6; nt++) {
        int col = h * CDh + nt * 16 + l15;
#pragma unroll
        for (int r = 0; r < 8; r++) {
            int row = q0 + wv * 16 + hi * 8 + r;
            float invl = (lrun[r] > 0.f) ? 1.f / lrun[r] : 0.f;
            CtxB[((size_t)b * CS + row) * CD + col] = f2bf(ctx[nt][r] * invl);
        }
    }
}

// ---------------------------------------------------------------- layernorm (one block per row)
__global__ __launch_bounds__(256)
void k_layernorm(const float* __restrict__ X, const float* __restrict__ g,
                 const float* __restrict__ be, float* __restrict__ Y,
                 u16* __restrict__ Yb)
{
    const int tid = threadIdx.x;
    const size_t row = blockIdx.x;
    const float* x = X + row * CD;
    float v0 = x[tid], v1 = x[tid + 256], v2 = x[tid + 512];
    float s  = v0 + v1 + v2;
    float ss = v0 * v0 + v1 * v1 + v2 * v2;
#pragma unroll
    for (int off = 1; off < 32; off <<= 1) {
        s  += __shfl_xor(s,  off, 32);
        ss += __shfl_xor(ss, off, 32);
    }
    __shared__ float sb[8], ssb[8];
    if ((tid & 31) == 0) { sb[tid >> 5] = s; ssb[tid >> 5] = ss; }
    __syncthreads();
    float ts = 0.f, tss = 0.f;
#pragma unroll
    for (int i = 0; i < 8; i++) { ts += sb[i]; tss += ssb[i]; }
    const float mu = ts * (1.f / CD);
    float var = tss * (1.f / CD) - mu * mu;
    var = var < 0.f ? 0.f : var;
    const float rs = rsqrtf(var + 1e-5f);
#pragma unroll
    for (int j = 0; j < 3; j++) {
        int c = tid + j * 256;
        float xv = (j == 0) ? v0 : ((j == 1) ? v1 : v2);
        float y = (xv - mu) * rs * g[c] + be[c];
        Y[row * CD + c] = y;
        if (Yb) Yb[row * CD + c] = f2bf(y);
    }
}

// ---------------------------------------------------------------- launcher
extern "C" void kernel_launch(void* const* d_in, const int* in_sizes, int n_in,
                              void* d_out, int out_size, void* d_ws, size_t ws_size,
                              hipStream_t stream)
{
    (void)in_sizes; (void)n_in; (void)out_size; (void)ws_size;
    const float* src    = (const float*)d_in[0];
    const float* coords = (const float*)d_in[1];
    const float* Wq = (const float*)d_in[2];   const float* bq  = (const float*)d_in[3];
    const float* Wk = (const float*)d_in[4];   const float* bk  = (const float*)d_in[5];
    const float* Wv = (const float*)d_in[6];   const float* bv  = (const float*)d_in[7];
    const float* Wo = (const float*)d_in[8];   const float* bo  = (const float*)d_in[9];
    const float* W1 = (const float*)d_in[10];  const float* b1  = (const float*)d_in[11];
    const float* W2 = (const float*)d_in[12];  const float* b2  = (const float*)d_in[13];
    const float* g1 = (const float*)d_in[14];  const float* be1 = (const float*)d_in[15];
    const float* g2 = (const float*)d_in[16];  const float* be2 = (const float*)d_in[17];
    float* out = (float*)d_out;

    char* w = (char*)d_ws;
    size_t off = 0;
    auto alloc = [&](size_t bytes) -> char* {
        char* p = w + off;
        off += (bytes + 255) & ~(size_t)255;
        return p;
    };
    const size_t fr  = (size_t)CNT * CD * sizeof(float);   // fp32 row matrix
    const size_t br  = (size_t)CNT * CD * sizeof(u16);     // bf16 row matrix
    const size_t bff = (size_t)CNT * CF * sizeof(u16);     // bf16 ffn matrix

    // persistent bf16 operands
    u16* srcB = (u16*)alloc(br);
    u16* WqB  = (u16*)alloc((size_t)CD * CD * 2);
    u16* WkB  = (u16*)alloc((size_t)CD * CD * 2);
    u16* WvB  = (u16*)alloc((size_t)CD * CD * 2);
    u16* WoB  = (u16*)alloc((size_t)CD * CD * 2);
    u16* W1B  = (u16*)alloc((size_t)CF * CD * 2);
    u16* W2B  = (u16*)alloc((size_t)CD * CF * 2);
    // arena A: three fp32 row matrices (reused)
    float* qraw = (float*)alloc(fr);
    float* kraw = (float*)alloc(fr);
    float* vraw = (float*)alloc(fr);
    // arena B: Qh/Kh/Vt (3 x br) then reused for h1B (bff > 3*br)
    char* arenaB = alloc(bff > 3 * br ? bff : 3 * br);
    u16* Qh = (u16*)arenaB;
    u16* Kh = Qh + (size_t)CNT * CD;
    u16* Vt = Kh + (size_t)CNT * CD;
    // stage-2 aliases (arena A)
    u16*   ctxB     = (u16*)qraw;
    float* attn_res = kraw;
    float* src2     = vraw;
    u16*   src2B    = (u16*)qraw;      // after ctxB consumed by O-proj
    float* ffn_out  = kraw;            // after attn_res consumed by LN1
    u16*   h1B      = (u16*)arenaB;    // after attention consumed Qh/Kh/Vt

    auto cvt = [&](const float* s_, u16* dst, size_t n) {
        k_f32_to_bf16<<<(unsigned)((n + 255) / 256), 256, 0, stream>>>(s_, dst, (int)n);
    };
    cvt(src, srcB, (size_t)CNT * CD);
    cvt(Wq, WqB, (size_t)CD * CD);
    cvt(Wk, WkB, (size_t)CD * CD);
    cvt(Wv, WvB, (size_t)CD * CD);
    cvt(Wo, WoB, (size_t)CD * CD);
    cvt(W1, W1B, (size_t)CF * CD);
    cvt(W2, W2B, (size_t)CD * CF);

    dim3 gD(CD / 128, CNT / 128);   // (6, 64)
    dim3 gF(CF / 128, CNT / 128);   // (24, 64)

    // QKV projections
    k_gemm<0><<<gD, 256, 0, stream>>>(srcB, WqB, bq, nullptr, qraw, nullptr, CNT, CD, CD);
    k_gemm<0><<<gD, 256, 0, stream>>>(srcB, WkB, bk, nullptr, kraw, nullptr, CNT, CD, CD);
    k_gemm<0><<<gD, 256, 0, stream>>>(srcB, WvB, bv, nullptr, vraw, nullptr, CNT, CD, CD);

    // RoPE + head reshape
    {
        size_t nr = (size_t)CNT * CH * 48;
        k_rope<<<(unsigned)((nr + 255) / 256), 256, 0, stream>>>(qraw, kraw, coords, Qh, Kh);
        size_t nv = (size_t)CNT * CD;
        k_vre<<<(unsigned)((nv + 255) / 256), 256, 0, stream>>>(vraw, Vt);
    }

    // attention
    k_attn<<<dim3(CS / 64, CB * CH), 128, 0, stream>>>(Qh, Kh, Vt, ctxB, 0.102062072615966f);

    // O projection + residual, LN1
    k_gemm<2><<<gD, 256, 0, stream>>>(ctxB, WoB, bo, src, attn_res, nullptr, CNT, CD, CD);
    k_layernorm<<<CNT, 256, 0, stream>>>(attn_res, g1, be1, src2, src2B);

    // FFN
    k_gemm<1><<<gF, 256, 0, stream>>>(src2B, W1B, b1, nullptr, nullptr, h1B, CNT, CF, CD);
    k_gemm<2><<<gD, 256, 0, stream>>>(h1B, W2B, b2, src2, ffn_out, nullptr, CNT, CD, CF);

    // LN2 -> output
    k_layernorm<<<CNT, 256, 0, stream>>>(ffn_out, g2, be2, out, nullptr);
}
